// MessagePassing_68298569941222
// MI455X (gfx1250) — matrix operand order repaired and checked by reading, not compiled, not run on hardware
//
#include <hip/hip_runtime.h>

// MI455X / gfx1250 message-passing GNN.
// msg = X @ W2 with X[e, b*32+j] = bond[e,b]*nbr[e,j]; bias folded in as bond
// feature #16 == 1.0.  Computed with v_wmma_f32_16x16x32_bf16 (fp32 accum):
// each wave handles 32 edges (two 16-edge M-tiles), 4 WMMAs per bond feature
// sharing one pair of LDS B-fragments -> 68 WMMAs per wave.
// A-fragments built with v_pk_mul_f32 + hardware v_cvt_pk_bf16_f32.
// Segment-sum via global_atomic_add_f32 (agg is L2-resident: whole working set
// ~35 MB << 192 MB L2, so HBM traffic is cold-miss only).  GRU in fp32 VALU
// with v4f (ds_load_b128) LDS weight reads.
//
// d_ws layout: [ h ping buffer: N*32 f32 (6.4 MB) ][ Bpack: 17*2*32 v16bf (34 KB) ]

typedef __attribute__((ext_vector_type(16))) __bf16 v16bf;
typedef __attribute__((ext_vector_type(2)))  __bf16 v2bf;
typedef __attribute__((ext_vector_type(8)))  float  v8f;
typedef __attribute__((ext_vector_type(4)))  float  v4f;

#define D 32
#define BD 16
#define STEPS 4
#define TILE 16            // edges per WMMA M-tile
#define WAVES_PER_BLOCK 8
#define NB 17              // BD + 1 (bias folded in)

// Packed f32->bf16 conversion: hardware op when available.
__device__ __forceinline__ v2bf pk2bf(float lo, float hi) {
#if __has_builtin(__builtin_amdgcn_cvt_pk_bf16_f32)
  auto t = __builtin_amdgcn_cvt_pk_bf16_f32(lo, hi);
  return __builtin_bit_cast(v2bf, t);
#else
  v2bf r; r[0] = (__bf16)lo; r[1] = (__bf16)hi; return r;
#endif
}

struct ABuf { v2bf p[8]; };

// Build one 16-bit A-fragment: a[p] = bf16(s * nb[p]).
__device__ __forceinline__ v16bf build_a(float s, const float (&nb)[16]) {
  ABuf t;
#pragma unroll
  for (int p = 0; p < 8; ++p) t.p[p] = pk2bf(s * nb[2 * p], s * nb[2 * p + 1]);
  return __builtin_bit_cast(v16bf, t);
}

// ---------------------------------------------------------------------------
// Repack W (16,32,32) + Bm (32,32) into per-lane bf16 B-fragments.
// B (KxN = 32x16) layout: lanes 0-15 hold K={0..7,16..23}, lanes 16-31 hold
// K={8..15,24..31} (matches A); N = lane&15.   B[k][n] = W[b][n][k].
// Bpack index: (b*2 + nhalf)*32 + lane;  nhalf = output features 0-15 / 16-31.
// ---------------------------------------------------------------------------
__global__ void pack_weights_kernel(const float* __restrict__ kern,
                                    const float* __restrict__ bias,
                                    v16bf* __restrict__ Bpack)
{
  const int idx = blockIdx.x * blockDim.x + threadIdx.x;
  if (idx >= NB * 2 * 32) return;
  const int lane = idx & 31;
  const int nh   = (idx >> 5) & 1;
  const int b    = idx >> 6;
  const int n    = nh * 16 + (lane & 15);       // output feature i
  const int kbase = (lane < 16) ? 0 : 8;        // j base for this lane half
  const float* Wb = (b < BD) ? (kern + (size_t)b * D * D) : bias;
  ABuf t;
#pragma unroll
  for (int r = 0; r < 8; ++r)
    t.p[r] = pk2bf(Wb[n * D + kbase + (r >> 2) * 16 + 2 * (r & 3) + 0], Wb[n * D + kbase + (r >> 2) * 16 + 2 * (r & 3) + 1]);
  Bpack[idx] = __builtin_bit_cast(v16bf, t);
}

__global__ void zero_kernel(float* __restrict__ p, long n) {
  long i = (long)blockIdx.x * blockDim.x + threadIdx.x;
  if (i < n) p[i] = 0.0f;
}

// ---------------------------------------------------------------------------
// Edge kernel: one wave32 computes msg for TWO 16-edge tiles and scatters.
// A (16-bit 16x32) layout: lanes 0-15 carry K={0..7,16..23}, lanes 16-31
// carry K={8..15,24..31}; M = lane&15.  A[m,j] = bond[m,b]*nbr[m,j].
// ---------------------------------------------------------------------------
__global__ void __launch_bounds__(256)
edge_msg_kernel(const float* __restrict__ hsrc,
                const float* __restrict__ bond,
                const int*   __restrict__ pairs,
                const v16bf* __restrict__ Bpack,
                float*       __restrict__ agg,
                int ntiles)
{
  __shared__ v16bf sBW[NB * 2 * 32];                      // 34816 B
  for (int i = threadIdx.x; i < NB * 2 * 32; i += blockDim.x) sBW[i] = Bpack[i];
  __syncthreads();

  const int lane = threadIdx.x & 31;
  const int wave = threadIdx.x >> 5;
  const int pairIdx = blockIdx.x * WAVES_PER_BLOCK + wave;
  const int t0 = pairIdx * 2;
  if (t0 >= ntiles) return;                               // wave-uniform
  const bool has1 = (t0 + 1) < ntiles;
  const int t1 = has1 ? (t0 + 1) : t0;

  const int m  = lane & 15;
  const int jb = (lane >> 4) * 8;                          // 0 or 8

  // ---- tile 0 operands ----
  const int eA = t0 * TILE + m;
  const int dstA = pairs[2 * eA + 0];
  const int srcA = pairs[2 * eA + 1];
  const float* hrA = hsrc + (size_t)srcA * D;
  v4f a0v = *(const v4f*)(hrA + jb + 0);
  v4f a1v = *(const v4f*)(hrA + jb + 4);
  v4f a2v = *(const v4f*)(hrA + jb + 16);
  v4f a3v = *(const v4f*)(hrA + jb + 20);
  float nbA[16] = { a0v.x, a0v.y, a0v.z, a0v.w, a1v.x, a1v.y, a1v.z, a1v.w,
                    a2v.x, a2v.y, a2v.z, a2v.w, a3v.x, a3v.y, a3v.z, a3v.w };
  const float* brA = bond + (size_t)eA * BD;
  v4f c0 = *(const v4f*)(brA + 0);
  v4f c1 = *(const v4f*)(brA + 4);
  v4f c2 = *(const v4f*)(brA + 8);
  v4f c3 = *(const v4f*)(brA + 12);
  float bdA[NB] = { c0.x, c0.y, c0.z, c0.w, c1.x, c1.y, c1.z, c1.w,
                    c2.x, c2.y, c2.z, c2.w, c3.x, c3.y, c3.z, c3.w, 1.0f };

  // ---- tile 1 operands (clamped to tile 0 when absent) ----
  const int eB = t1 * TILE + m;
  const int dstB = pairs[2 * eB + 0];
  const int srcB = pairs[2 * eB + 1];
  const float* hrB = hsrc + (size_t)srcB * D;
  v4f b0v = *(const v4f*)(hrB + jb + 0);
  v4f b1v = *(const v4f*)(hrB + jb + 4);
  v4f b2v = *(const v4f*)(hrB + jb + 16);
  v4f b3v = *(const v4f*)(hrB + jb + 20);
  float nbB[16] = { b0v.x, b0v.y, b0v.z, b0v.w, b1v.x, b1v.y, b1v.z, b1v.w,
                    b2v.x, b2v.y, b2v.z, b2v.w, b3v.x, b3v.y, b3v.z, b3v.w };
  const float* brB = bond + (size_t)eB * BD;
  v4f d0 = *(const v4f*)(brB + 0);
  v4f d1 = *(const v4f*)(brB + 4);
  v4f d2 = *(const v4f*)(brB + 8);
  v4f d3 = *(const v4f*)(brB + 12);
  float bdB[NB] = { d0.x, d0.y, d0.z, d0.w, d1.x, d1.y, d1.z, d1.w,
                    d2.x, d2.y, d2.z, d2.w, d3.x, d3.y, d3.z, d3.w, 1.0f };

  v8f acc00 = {};   // tile0, features 0..15
  v8f acc01 = {};   // tile0, features 16..31
  v8f acc10 = {};   // tile1, features 0..15
  v8f acc11 = {};   // tile1, features 16..31

#pragma unroll
  for (int b = 0; b < NB; ++b) {
    v16bf a0 = build_a(bdA[b], nbA);
    v16bf a1 = build_a(bdB[b], nbB);
    v16bf w0 = sBW[(b * 2 + 0) * 32 + lane];
    v16bf w1 = sBW[(b * 2 + 1) * 32 + lane];
    // Round-robin over 4 independent accumulator chains to space out the
    // bf16 WMMA->WMMA D->C hazard (5 slots) with co-executable VALU.
    acc00 = __builtin_amdgcn_wmma_f32_16x16x32_bf16(false, a0, false, w0,
                                                    (short)0, acc00, false, false);
    acc10 = __builtin_amdgcn_wmma_f32_16x16x32_bf16(false, a1, false, w0,
                                                    (short)0, acc10, false, false);
    acc01 = __builtin_amdgcn_wmma_f32_16x16x32_bf16(false, a0, false, w1,
                                                    (short)0, acc01, false, false);
    acc11 = __builtin_amdgcn_wmma_f32_16x16x32_bf16(false, a1, false, w1,
                                                    (short)0, acc11, false, false);
  }

  // D layout: lane L holds N = L&15; rows M = (L>>4)*8 + r live in VGPR r.
  const int n  = lane & 15;
  const int mh = (lane >> 4) * 8;
#pragma unroll
  for (int r = 0; r < 8; ++r) {
    const int dA = __shfl(dstA, mh + r, 32);   // dst of edge (t0*16 + mh + r)
    float* row = agg + (size_t)dA * D;
    atomicAdd(row + n,      acc00[r]);
    atomicAdd(row + 16 + n, acc01[r]);
  }
  if (has1) {
#pragma unroll
    for (int r = 0; r < 8; ++r) {
      const int dB = __shfl(dstB, mh + r, 32);
      float* row = agg + (size_t)dB * D;
      atomicAdd(row + n,      acc10[r]);
      atomicAdd(row + 16 + n, acc11[r]);
    }
  }
}

// ---------------------------------------------------------------------------
// GRU cell, fp32, one thread per atom.  Weights cached in LDS as v4f so every
// weight read is a ds_load_b128.  xv/hv live purely in registers (only ever
// indexed by unrolled constants); the single dynamically-indexed blend scalar
// h[i] is re-read from the global row (L0/L2-hot) to avoid scratch spills.
// Safe in-place (row fully read before any write of the same row).
// ---------------------------------------------------------------------------
__device__ __forceinline__ float sigmoidf_(float x) { return 1.0f / (1.0f + __expf(-x)); }

#define DV (D / 4)   // v4f chunks per row

__global__ void __launch_bounds__(256)
gru_kernel(const float* __restrict__ x, const float* __restrict__ hin,
           float* __restrict__ hout,
           const float* __restrict__ Wih, const float* __restrict__ Whh,
           const float* __restrict__ bih, const float* __restrict__ bhh,
           int n)
{
  __shared__ v4f sWih[3 * D * DV];
  __shared__ v4f sWhh[3 * D * DV];
  __shared__ float sbih[3 * D];
  __shared__ float sbhh[3 * D];
  const v4f* gWih = (const v4f*)Wih;
  const v4f* gWhh = (const v4f*)Whh;
  for (int i = threadIdx.x; i < 3 * D * DV; i += blockDim.x) { sWih[i] = gWih[i]; sWhh[i] = gWhh[i]; }
  for (int i = threadIdx.x; i < 3 * D; i += blockDim.x)      { sbih[i] = bih[i]; sbhh[i] = bhh[i]; }
  __syncthreads();

  const int atom = blockIdx.x * blockDim.x + threadIdx.x;
  if (atom >= n) return;

  const float* hrow_f = hin + (size_t)atom * D;   // scalar re-reads for blend
  v4f xv[DV], hv[DV];
  const v4f* xr = (const v4f*)(x + (size_t)atom * D);
  const v4f* hr = (const v4f*)hrow_f;
#pragma unroll
  for (int c = 0; c < DV; ++c) { xv[c] = xr[c]; hv[c] = hr[c]; }

  float* orow = hout + (size_t)atom * D;
  for (int i = 0; i < D; ++i) {
    const v4f* wir = sWih + i * DV;
    const v4f* whr = sWhh + i * DV;
    const v4f* wiz = sWih + (D + i) * DV;
    const v4f* whz = sWhh + (D + i) * DV;
    const v4f* win = sWih + (2 * D + i) * DV;
    const v4f* whn = sWhh + (2 * D + i) * DV;
    float gir = sbih[i],         ghr = sbhh[i];
    float giz = sbih[D + i],     ghz = sbhh[D + i];
    float gin = sbih[2 * D + i], ghn = sbhh[2 * D + i];
    const float hi_s = hrow_f[i];                 // global, not register array
#pragma unroll
    for (int c = 0; c < DV; ++c) {
      const v4f xc = xv[c], hc = hv[c];
      const v4f w0 = wir[c], w1 = whr[c], w2 = wiz[c];
      const v4f w3 = whz[c], w4 = win[c], w5 = whn[c];
      gir = fmaf(w0.x, xc.x, gir); gir = fmaf(w0.y, xc.y, gir);
      gir = fmaf(w0.z, xc.z, gir); gir = fmaf(w0.w, xc.w, gir);
      ghr = fmaf(w1.x, hc.x, ghr); ghr = fmaf(w1.y, hc.y, ghr);
      ghr = fmaf(w1.z, hc.z, ghr); ghr = fmaf(w1.w, hc.w, ghr);
      giz = fmaf(w2.x, xc.x, giz); giz = fmaf(w2.y, xc.y, giz);
      giz = fmaf(w2.z, xc.z, giz); giz = fmaf(w2.w, xc.w, giz);
      ghz = fmaf(w3.x, hc.x, ghz); ghz = fmaf(w3.y, hc.y, ghz);
      ghz = fmaf(w3.z, hc.z, ghz); ghz = fmaf(w3.w, hc.w, ghz);
      gin = fmaf(w4.x, xc.x, gin); gin = fmaf(w4.y, xc.y, gin);
      gin = fmaf(w4.z, xc.z, gin); gin = fmaf(w4.w, xc.w, gin);
      ghn = fmaf(w5.x, hc.x, ghn); ghn = fmaf(w5.y, hc.y, ghn);
      ghn = fmaf(w5.z, hc.z, ghn); ghn = fmaf(w5.w, hc.w, ghn);
    }
    float rg = sigmoidf_(gir + ghr);
    float zg = sigmoidf_(giz + ghz);
    float ng = tanhf(gin + rg * ghn);
    orow[i] = (1.0f - zg) * ng + zg * hi_s;
  }
}

// ---------------------------------------------------------------------------
extern "C" void kernel_launch(void* const* d_in, const int* in_sizes, int n_in,
                              void* d_out, int out_size, void* d_ws, size_t ws_size,
                              hipStream_t stream)
{
  const float* atom = (const float*)d_in[0];
  const float* bond = (const float*)d_in[1];
  const int*   pairs= (const int*)  d_in[2];
  const float* kern = (const float*)d_in[3];
  const float* bias = (const float*)d_in[4];
  const float* Wih  = (const float*)d_in[5];
  const float* Whh  = (const float*)d_in[6];
  const float* bih  = (const float*)d_in[7];
  const float* bhh  = (const float*)d_in[8];

  const int N = in_sizes[0] / D;
  const int E = in_sizes[2] / 2;
  const int ntiles = E / TILE;              // E = 300000 is a multiple of 16
  const int npairs = (ntiles + 1) / 2;

  float* out  = (float*)d_out;              // doubles as the agg buffer
  float* hbuf = (float*)d_ws;               // h ping buffer
  v16bf* Bpack = (v16bf*)((char*)d_ws + (size_t)N * D * sizeof(float));

  pack_weights_kernel<<<(NB * 2 * 32 + 255) / 256, 256, 0, stream>>>(kern, bias, Bpack);

  for (int step = 0; step < STEPS; ++step) {
    const float* hcur = (step == 0) ? atom : hbuf;
    zero_kernel<<<(unsigned)(((long)N * D + 255) / 256), 256, 0, stream>>>(out, (long)N * D);
    edge_msg_kernel<<<(npairs + WAVES_PER_BLOCK - 1) / WAVES_PER_BLOCK, 256, 0, stream>>>(
        hcur, bond, pairs, Bpack, out, ntiles);
    float* hnext = (step == STEPS - 1) ? out : hbuf;   // final step writes d_out
    gru_kernel<<<(N + 255) / 256, 256, 0, stream>>>(out, hcur, hnext,
                                                    Wih, Whh, bih, bhh, N);
  }
}
